// PHD_DecoderBlock_25220047962477
// MI455X (gfx1250) — matrix operand order repaired and checked by reading, not compiled, not run on hardware
//
#include <hip/hip_runtime.h>
#include <hip/hip_bf16.h>
#include <cstdint>
#include <cstddef>

// ---------------------------------------------------------------------------
// PHD DecoderBlock for MI455X (gfx1250, wave32, WMMA).
// All GEMMs run through v_wmma_f32_16x16x32_bf16; fragments are loaded
// directly from global memory in a contiguous-k lane layout (k = 16*half + e, same for A and B).
// Tile-validity is wave-uniform and scalarized via readfirstlane so the
// K-loop has no EXEC manipulation.
// The selective scan is fused (dt-proj + softplus + exp + state + gate).
// Workspace requirement: ~71 MB.
// ---------------------------------------------------------------------------

typedef __bf16 bf16;
typedef __attribute__((ext_vector_type(16))) __bf16 v16bf;
typedef __attribute__((ext_vector_type(8)))  __bf16 v8bf;
typedef __attribute__((ext_vector_type(8)))  float  v8f;

// Problem dimensions (fixed by setup_inputs()).
constexpr int Bb   = 2;
constexpr int Cin  = 512;
constexpr int Cc   = 256;
constexpr int Hh   = 48;
constexpr int Ww   = 48;
constexpr int HW   = Hh * Ww;        // 2304 == L
constexpr int Ls   = HW;
constexpr int Mrows= Bb * Ls;        // 4608
constexpr int DI   = 512;
constexpr int DS   = 16;
constexpr int DTR  = 16;
constexpr float EPSV = 1e-5f;

// RNE float -> bf16 (bit twiddling; avoids relying on cast lowering).
__device__ inline bf16 to_bf16(float f) {
  unsigned u = __builtin_bit_cast(unsigned, f);
  unsigned r = (u + 0x7FFFu + ((u >> 16) & 1u)) >> 16;
  unsigned short h = (unsigned short)r;
  return __builtin_bit_cast(bf16, h);
}

__device__ inline float softplusf(float x) {
  return (x > 20.f) ? x : log1pf(__expf(x));
}

// Sequence-position -> spatial-position map for the 4 scan directions.
__device__ inline int map_dir(int dir, int t) {
  switch (dir) {
    case 0:  return t;
    case 1:  return Ls - 1 - t;
    case 2:  { int h = t % Hh; int w = t / Hh; return h * Ww + w; }
    default: { int tt = Ls - 1 - t; int h = tt % Hh; int w = tt / Hh; return h * Ww + w; }
  }
}

__device__ inline float block_sum_256(float v, float* red) {
  const int c = threadIdx.x;
  red[c] = v;
  __syncthreads();
  #pragma unroll
  for (int o = 128; o > 0; o >>= 1) {
    if (c < o) red[c] += red[c + o];
    __syncthreads();
  }
  float r = red[0];
  __syncthreads();
  return r;
}

// ---------------------------------------------------------------------------
// WMMA GEMM body, NT valid 16-col tiles (compile-time).  All N used here are
// multiples of 16, so every selected tile is full -> no epilogue guards.
// ---------------------------------------------------------------------------
template<int EP, bool WBF, int NT>
__device__ inline void gemm_body(const bf16* __restrict__ aRow,
                                 const bf16* __restrict__ Bw,
                                 float* __restrict__ O, bf16* __restrict__ Obf,
                                 const float* __restrict__ gvec,
                                 const float* __restrict__ bias,
                                 int m0, int n0, int half, int l15,
                                 int K, int ldo, int dir)
{
  v8f acc[NT];
  #pragma unroll
  for (int j = 0; j < NT; ++j) acc[j] = v8f{};

  // Per-tile B column base: lane holds col n; elems e -> K = k + half*16 + e.
  const bf16* bCol[NT];
  #pragma unroll
  for (int j = 0; j < NT; ++j)
    bCol[j] = Bw + (size_t)(n0 + j * 16 + l15) * K + half * 16;

  for (int k = 0; k < K; k += 32) {
    // A fragment: lane holds row (lane&15); elems e -> K=k+half*16+e
    // (one 32B load, matching B).
    union { v16bf v; v8bf h[2]; } a;
    a.v = *(const v16bf*)(aRow + k + half * 16);
    __builtin_prefetch(aRow + k + 64, 0, 3);   // global_prefetch_b8

    #pragma unroll
    for (int j = 0; j < NT; ++j) {
      v16bf bfrag = *(const v16bf*)(bCol[j] + k);
      acc[j] = __builtin_amdgcn_wmma_f32_16x16x32_bf16(
                   false, a.v, false, bfrag, (short)0, acc[j], false, false);
    }
  }

  const float rsq = rsqrtf(1.0f + EPSV);
  #pragma unroll
  for (int j = 0; j < NT; ++j) {
    const int n = n0 + j * 16 + l15;
    #pragma unroll
    for (int r = 0; r < 8; ++r) {
      const int m = m0 + half * 8 + r;   // C/D layout: VGPR r -> row r (+8 hi half)
      float v = acc[j][r];
      if (EP == 1) {                     // BN + ReLU
        v = fmaxf(v * (gvec[n] * rsq) + bias[n], 0.f);
        O[(size_t)m * ldo + n] = v;
        if (WBF) Obf[(size_t)m * ldo + n] = to_bf16(v);
      } else if (EP == 2) {              // + bias
        O[(size_t)m * ldo + n] = v + bias[n];
      } else if (EP == 3) {              // scatter-add 0.25*v via direction map
        const int b  = m / Ls;
        const int t  = m - b * Ls;
        const int sp = map_dir(dir, t);
        O[((size_t)b * Ls + sp) * ldo + n] += 0.25f * v;
      } else {                           // plain f32 store
        O[(size_t)m * ldo + n] = v;
      }
    }
  }
}

// Generic WMMA GEMM:  O[M,N] = A[M,K](bf16,row-major) @ W[N,K](bf16,row-major)^T
// One wave -> one 16-row tile x up to four 16-col tiles (A fragment reused).
template<int EP, bool WBF>
__global__ __launch_bounds__(256)
void wmma_gemm(const bf16* __restrict__ A, const bf16* __restrict__ Bw,
               float* __restrict__ O, bf16* __restrict__ Obf,
               const float* __restrict__ gvec, const float* __restrict__ bias,
               int M, int N, int K, int ldo, int dir)
{
  const int lane = threadIdx.x & 31;
  const int wave = (int)((blockIdx.x * blockDim.x + threadIdx.x) >> 5);
  const int nT4  = (N + 63) >> 6;
  const int mT   = M >> 4;
  if (wave >= mT * nT4) return;
  const int m0   = (wave / nT4) * 16;
  const int n0   = (wave % nT4) * 64;
  const int half = lane >> 4;
  const int l15  = lane & 15;

  const bf16* aRow = A + (size_t)(m0 + l15) * K;

  // Number of valid 16-col tiles for this wave; wave-uniform -> scalarize.
  int nt = (N - n0) >> 4;
  if (nt > 4) nt = 4;
  nt = __builtin_amdgcn_readfirstlane(nt);

  switch (nt) {
    case 4:
      gemm_body<EP, WBF, 4>(aRow, Bw, O, Obf, gvec, bias, m0, n0, half, l15, K, ldo, dir);
      break;
    case 3:
      gemm_body<EP, WBF, 3>(aRow, Bw, O, Obf, gvec, bias, m0, n0, half, l15, K, ldo, dir);
      break;
    case 2:
      gemm_body<EP, WBF, 2>(aRow, Bw, O, Obf, gvec, bias, m0, n0, half, l15, K, ldo, dir);
      break;
    default:
      gemm_body<EP, WBF, 1>(aRow, Bw, O, Obf, gvec, bias, m0, n0, half, l15, K, ldo, dir);
      break;
  }
}

// ---------------------------------------------------------------------------
// Utility kernels
// ---------------------------------------------------------------------------
__global__ void k_f32_to_bf16(const float* __restrict__ in, bf16* __restrict__ out, int n) {
  int i = blockIdx.x * blockDim.x + threadIdx.x;
  if (i < n) out[i] = to_bf16(in[i]);
}

__global__ void k_copy_f32(const float* __restrict__ in, float* __restrict__ out, int n) {
  int i = blockIdx.x * blockDim.x + threadIdx.x;
  if (i < n) out[i] = in[i];
}

// x NCHW f32 -> NHWC bf16
__global__ void k_x_to_bf16(const float* __restrict__ x, bf16* __restrict__ o) {
  int i = blockIdx.x * blockDim.x + threadIdx.x;
  if (i >= Bb * HW * Cin) return;
  int cin = i % Cin;
  int hw  = (i / Cin) % HW;
  int b   = i / (Cin * HW);
  o[i] = to_bf16(x[((size_t)b * Cin + cin) * HW + hw]);
}

// Depthwise 1x7 and 7x1 convs + BN + ReLU, summed -> bf16 (NHWC).
__global__ void k_dwconv_hv(const float* __restrict__ p,
                            const float* __restrict__ dwh, const float* __restrict__ dwv,
                            const float* __restrict__ gh,  const float* __restrict__ bh,
                            const float* __restrict__ gv,  const float* __restrict__ bv,
                            bf16* __restrict__ hv)
{
  int i = blockIdx.x * blockDim.x + threadIdx.x;
  if (i >= Bb * HW * Cc) return;
  int c  = i % Cc;
  int hw = (i / Cc) % HW;
  int b  = i / (Cc * HW);
  int hr = hw / Ww, w = hw % Ww;
  float ah = 0.f, av = 0.f;
  #pragma unroll
  for (int j = -3; j <= 3; ++j) {
    int wj = w + j;
    if (wj >= 0 && wj < Ww)
      ah += p[(((size_t)b * HW) + hr * Ww + wj) * Cc + c] * dwh[c * 7 + (j + 3)];
    int hj = hr + j;
    if (hj >= 0 && hj < Hh)
      av += p[(((size_t)b * HW) + hj * Ww + w) * Cc + c] * dwv[c * 7 + (j + 3)];
  }
  const float rsq = rsqrtf(1.0f + EPSV);
  float vh = fmaxf(ah * (gh[c] * rsq) + bh[c], 0.f);
  float vv = fmaxf(av * (gv[c] * rsq) + bv[c], 0.f);
  hv[i] = to_bf16(vh + vv);
}

// LayerNorm over C with direction-mapped gather; one block per token.
__global__ __launch_bounds__(256)
void k_ln_gather(const float* __restrict__ xr,
                 const float* __restrict__ g, const float* __restrict__ bvec,
                 bf16* __restrict__ xn, int dir)
{
  __shared__ float red[256];
  const int token = blockIdx.x;               // 0 .. B*L-1
  const int b = token / Ls, t = token - b * Ls;
  const int s = map_dir(dir, t);
  const int c = threadIdx.x;
  const float v  = xr[((size_t)b * Ls + s) * Cc + c];
  const float mu = block_sum_256(v, red) * (1.f / Cc);
  const float dv = v - mu;
  const float va = block_sum_256(dv * dv, red) * (1.f / Cc);
  const float o  = dv * rsqrtf(va + EPSV) * g[c] + bvec[c];
  xn[(size_t)token * Cc + c] = to_bf16(o);
}

// Causal depthwise conv1d (k=4) along the sequence + SiLU.
__global__ void k_conv_silu(const float* __restrict__ xz,
                            const float* __restrict__ cw, const float* __restrict__ cb,
                            float* __restrict__ xm, bf16* __restrict__ xmbf)
{
  int i = blockIdx.x * blockDim.x + threadIdx.x;
  if (i >= Bb * Ls * DI) return;
  int d  = i % DI;
  int bt = i / DI;
  int t  = bt % Ls;
  float acc = cb[d];
  #pragma unroll
  for (int j = 0; j < 4; ++j) {
    int tt = t - 3 + j;
    if (tt >= 0) acc += cw[d * 4 + j] * xz[(size_t)(bt - t + tt) * (2 * DI) + d];
  }
  float o = acc * (1.f / (1.f + __expf(-acc)));
  xm  [(size_t)bt * DI + d] = o;
  xmbf[(size_t)bt * DI + d] = to_bf16(o);
}

// Fused selective scan: dt-projection + softplus + exp recurrence +
// C contraction + D skip + silu(z) gate.  One block per batch, thread=channel.
__global__ __launch_bounds__(512)
void k_scan(const float* __restrict__ proj, const float* __restrict__ xm,
            const float* __restrict__ xz,
            const float* __restrict__ dtw, const float* __restrict__ dtb,
            const float* __restrict__ Alog, const float* __restrict__ Dp,
            bf16* __restrict__ ybf)
{
  constexpr int CH = 48;                      // steps per LDS chunk (2304/48)
  __shared__ float pbuf[CH * 48];
  const int b = blockIdx.x;
  const int d = threadIdx.x;

  float Ar[DS], Wr[DS], h[DS];
  #pragma unroll
  for (int s = 0; s < DS; ++s) {
    Ar[s] = -__expf(Alog[d * DS + s]);
    Wr[s] = dtw[d * DTR + s];
    h[s]  = 0.f;
  }
  const float dtb_r = dtb[d];
  const float Dp_r  = Dp[d];

  for (int c0 = 0; c0 < Ls; c0 += CH) {
    for (int i = d; i < CH * 48; i += DI)
      pbuf[i] = proj[((size_t)b * Ls + c0) * 48 + i];
    __syncthreads();

    for (int tc = 0; tc < CH; ++tc) {
      const int t = c0 + tc;
      const float* pr = pbuf + tc * 48;       // [dt(16) | B(16) | C(16)]
      float dt = dtb_r;
      #pragma unroll
      for (int s = 0; s < DTR; ++s) dt += pr[s] * Wr[s];
      dt = softplusf(dt);

      const float xv  = xm[((size_t)b * Ls + t) * DI + d];
      const float dbx = dt * xv;
      float y = 0.f;
      #pragma unroll
      for (int s = 0; s < DS; ++s) {
        h[s] = __expf(dt * Ar[s]) * h[s] + dbx * pr[16 + s];
        y   += h[s] * pr[32 + s];
      }
      y += Dp_r * xv;
      const float zv = xz[((size_t)b * Ls + t) * (2 * DI) + DI + d];
      y *= zv * (1.f / (1.f + __expf(-zv)));
      ybf[((size_t)b * Ls + t) * DI + d] = to_bf16(y);
    }
    __syncthreads();
  }
}

// s[b,c] = mean over HW of (feat_local + feat_global)
__global__ __launch_bounds__(256)
void k_mean_hw(const float* __restrict__ fl, const float* __restrict__ fg,
               float* __restrict__ s)
{
  __shared__ float red[256];
  const int bc = blockIdx.x;
  const int b = bc / Cc, c = bc % Cc;
  float acc = 0.f;
  for (int p = threadIdx.x; p < HW; p += 256)
    acc += fl[((size_t)b * HW + p) * Cc + c] + fg[((size_t)b * HW + p) * Cc + c];
  acc = block_sum_256(acc, red);
  if (threadIdx.x == 0) s[bc] = acc * (1.f / HW);
}

// Squeeze-excite MLP + 2-way softmax.  wgt layout: [B,2,C].
__global__ __launch_bounds__(256)
void k_se(const float* __restrict__ s, const float* __restrict__ fc1,
          const float* __restrict__ fc2, float* __restrict__ wgt)
{
  __shared__ float t1[Bb * 16];
  const int tid = threadIdx.x;
  if (tid < Bb * 16) {
    int b = tid >> 4, m = tid & 15;
    float a = 0.f;
    for (int c = 0; c < Cc; ++c) a += s[b * Cc + c] * fc1[m * Cc + c];
    t1[tid] = fmaxf(a, 0.f);
  }
  __syncthreads();
  for (int i = tid; i < Bb * Cc; i += 256) {
    int b = i / Cc, c = i % Cc;
    float z0 = 0.f, z1 = 0.f;
    #pragma unroll
    for (int m = 0; m < 16; ++m) {
      z0 += t1[b * 16 + m] * fc2[c * 16 + m];
      z1 += t1[b * 16 + m] * fc2[(Cc + c) * 16 + m];
    }
    float mx = fmaxf(z0, z1);
    float e0 = __expf(z0 - mx), e1 = __expf(z1 - mx);
    float w0 = e0 / (e0 + e1);
    wgt[b * 2 * Cc + c]      = w0;
    wgt[b * 2 * Cc + Cc + c] = 1.f - w0;
  }
}

// out NCHW = w0*feat_local + w1*feat_global   (NHWC -> NCHW)
__global__ void k_combine(const float* __restrict__ fl, const float* __restrict__ fg,
                          const float* __restrict__ wgt, float* __restrict__ out)
{
  int i = blockIdx.x * blockDim.x + threadIdx.x;
  if (i >= Bb * Cc * HW) return;
  int hw = i % HW;
  int c  = (i / HW) % Cc;
  int b  = i / (HW * Cc);
  float w0 = wgt[b * 2 * Cc + c];
  float w1 = wgt[b * 2 * Cc + Cc + c];
  out[i] = w0 * fl[((size_t)b * HW + hw) * Cc + c]
         + w1 * fg[((size_t)b * HW + hw) * Cc + c];
}

// ---------------------------------------------------------------------------
// Host orchestration
// ---------------------------------------------------------------------------
extern "C" void kernel_launch(void* const* d_in, const int* in_sizes, int n_in,
                              void* d_out, int out_size, void* d_ws, size_t ws_size,
                              hipStream_t stream) {
  (void)in_sizes; (void)n_in; (void)out_size; (void)ws_size;

  const float* x        = (const float*)d_in[0];
  const float* reduce_w = (const float*)d_in[1];
  const float* bn0_g    = (const float*)d_in[2];
  const float* bn0_b    = (const float*)d_in[3];
  const float* proj_w   = (const float*)d_in[4];
  const float* bn1_g    = (const float*)d_in[5];
  const float* bn1_b    = (const float*)d_in[6];
  const float* dwh_w    = (const float*)d_in[7];
  const float* bnh_g    = (const float*)d_in[8];
  const float* bnh_b    = (const float*)d_in[9];
  const float* dwv_w    = (const float*)d_in[10];
  const float* bnv_g    = (const float*)d_in[11];
  const float* bnv_b    = (const float*)d_in[12];
  const float* fus_w    = (const float*)d_in[13];
  const float* fus_b    = (const float*)d_in[14];
  const float* ln_g     = (const float*)d_in[15];
  const float* ln_b     = (const float*)d_in[16];
  const float* in_w     = (const float*)d_in[17];
  const float* conv_w   = (const float*)d_in[18];
  const float* conv_b   = (const float*)d_in[19];
  const float* xproj_w  = (const float*)d_in[20];
  const float* dt_w     = (const float*)d_in[21];
  const float* dt_b     = (const float*)d_in[22];
  const float* A_log    = (const float*)d_in[23];
  const float* Dp       = (const float*)d_in[24];
  const float* out_w    = (const float*)d_in[25];
  const float* fc1_w    = (const float*)d_in[26];
  const float* fc2_w    = (const float*)d_in[27];

  char*  ws  = (char*)d_ws;
  size_t off = 0;
  auto alloc = [&](size_t bytes) -> char* {
    char* p = ws + off;
    off += (bytes + 255) & ~(size_t)255;
    return p;
  };

  // bf16 weight copies ([N,K] row-major == WMMA B-fragment friendly)
  bf16*  w_reduce = (bf16*)alloc((size_t)Cc * Cin * 2);
  bf16*  w_proj   = (bf16*)alloc((size_t)Cc * Cc * 2);
  bf16*  w_fus    = (bf16*)alloc((size_t)Cc * Cc * 2);
  bf16*  w_in     = (bf16*)alloc((size_t)(2 * DI) * Cc * 2);
  bf16*  w_xproj  = (bf16*)alloc((size_t)48 * DI * 2);
  bf16*  w_out    = (bf16*)alloc((size_t)Cc * DI * 2);
  // activations
  bf16*  x_bf     = (bf16*)alloc((size_t)Mrows * Cin * 2);   // NHWC bf16 input
  float* x_r      = (float*)alloc((size_t)Mrows * Cc * 4);   // relu(bn0(reduce))
  bf16*  x_r_bf   = (bf16*)alloc((size_t)Mrows * Cc * 2);
  float* p_buf    = (float*)alloc((size_t)Mrows * Cc * 4);
  bf16*  hv_bf    = (bf16*)alloc((size_t)Mrows * Cc * 2);
  float* f_local  = (float*)alloc((size_t)Mrows * Cc * 4);
  float* f_global = (float*)alloc((size_t)Mrows * Cc * 4);
  bf16*  xn_bf    = (bf16*)alloc((size_t)Mrows * Cc * 2);
  float* xz       = (float*)alloc((size_t)Mrows * 2 * DI * 4);
  float* xm       = (float*)alloc((size_t)Mrows * DI * 4);
  bf16*  xm_bf    = (bf16*)alloc((size_t)Mrows * DI * 2);
  float* proj48   = (float*)alloc((size_t)Mrows * 48 * 4);
  bf16*  y_bf     = (bf16*)alloc((size_t)Mrows * DI * 2);
  float* s_buf    = (float*)alloc((size_t)Bb * Cc * 4);
  float* wgt      = (float*)alloc((size_t)Bb * 2 * Cc * 4);

  auto cvt = [&](const float* src, bf16* dst, int n) {
    k_f32_to_bf16<<<(n + 255) / 256, 256, 0, stream>>>(src, dst, n);
  };
  cvt(reduce_w, w_reduce, Cc * Cin);
  cvt(proj_w,   w_proj,   Cc * Cc);
  cvt(fus_w,    w_fus,    Cc * Cc);
  cvt(in_w,     w_in,     2 * DI * Cc);
  cvt(xproj_w,  w_xproj,  48 * DI);
  cvt(out_w,    w_out,    Cc * DI);

  k_x_to_bf16<<<(Bb * HW * Cin + 255) / 256, 256, 0, stream>>>(x, x_bf);

  auto gemmBlocks = [](int M, int N) {
    int waves = (M / 16) * ((N + 63) / 64);
    return (waves + 7) / 8;
  };

  // reduce: [4608,512] x [256,512]^T -> BN0+ReLU -> x_r (f32 + bf16)
  wmma_gemm<1, true><<<gemmBlocks(Mrows, Cc), 256, 0, stream>>>(
      x_bf, w_reduce, x_r, x_r_bf, bn0_g, bn0_b, Mrows, Cc, Cin, Cc, 0);

  // feat_global starts as x
  k_copy_f32<<<(Mrows * Cc + 255) / 256, 256, 0, stream>>>(x_r, f_global, Mrows * Cc);

  // proj: [4608,256] x [256,256]^T -> BN1+ReLU -> p (f32)
  wmma_gemm<1, false><<<gemmBlocks(Mrows, Cc), 256, 0, stream>>>(
      x_r_bf, w_proj, p_buf, nullptr, bn1_g, bn1_b, Mrows, Cc, Cc, Cc, 0);

  // depthwise 1x7 / 7x1 + BN + ReLU, summed
  k_dwconv_hv<<<(Bb * HW * Cc + 255) / 256, 256, 0, stream>>>(
      p_buf, dwh_w, dwv_w, bnh_g, bnh_b, bnv_g, bnv_b, hv_bf);

  // fus: (h+v) x fus_w^T + fus_b -> feat_local
  wmma_gemm<2, false><<<gemmBlocks(Mrows, Cc), 256, 0, stream>>>(
      hv_bf, w_fus, f_local, nullptr, nullptr, fus_b, Mrows, Cc, Cc, Cc, 0);

  // 4-direction Mamba scans
  for (int dir = 0; dir < 4; ++dir) {
    k_ln_gather<<<Mrows, 256, 0, stream>>>(x_r, ln_g, ln_b, xn_bf, dir);

    wmma_gemm<0, false><<<gemmBlocks(Mrows, 2 * DI), 256, 0, stream>>>(
        xn_bf, w_in, xz, nullptr, nullptr, nullptr, Mrows, 2 * DI, Cc, 2 * DI, 0);

    k_conv_silu<<<(Bb * Ls * DI + 255) / 256, 256, 0, stream>>>(
        xz, conv_w, conv_b, xm, xm_bf);

    wmma_gemm<0, false><<<gemmBlocks(Mrows, 48), 256, 0, stream>>>(
        xm_bf, w_xproj, proj48, nullptr, nullptr, nullptr, Mrows, 48, DI, 48, 0);

    k_scan<<<Bb, DI, 0, stream>>>(proj48, xm, xz, dt_w, dt_b, A_log, Dp, y_bf);

    wmma_gemm<3, false><<<gemmBlocks(Mrows, Cc), 256, 0, stream>>>(
        y_bf, w_out, f_global, nullptr, nullptr, nullptr, Mrows, Cc, DI, Cc, dir);
  }

  // squeeze-excite gate + final combine
  k_mean_hw<<<Bb * Cc, 256, 0, stream>>>(f_local, f_global, s_buf);
  k_se<<<1, 256, 0, stream>>>(s_buf, fc1_w, fc2_w, wgt);
  k_combine<<<(Bb * Cc * HW + 255) / 256, 256, 0, stream>>>(
      f_local, f_global, wgt, (float*)d_out);
}
